// TransformerLikexLSTMBlock_86603720556878
// MI455X (gfx1250) — matrix operand order repaired and checked by reading, not compiled, not run on hardware
//
#include <hip/hip_runtime.h>
#include <cstddef>
#include <cstdint>

// ---------------- problem dims ----------------
#define B_     4
#define S_     1024
#define D_     512
#define INNER_ 1024
#define NH_    4
#define DH_    256
#define ROWS_  (B_ * S_)          // 4096
#define MLP_H_ 2048

typedef __bf16 bf16_t;
typedef bf16_t v16bf __attribute__((ext_vector_type(16)));
typedef bf16_t v8bf  __attribute__((ext_vector_type(8)));
typedef float  v8f   __attribute__((ext_vector_type(8)));

__device__ __forceinline__ float silu_f(float x) { return x / (1.0f + __expf(-x)); }

__device__ __forceinline__ v16bf join16(v8bf lo, v8bf hi) {
  v16bf r;
  #pragma unroll
  for (int j = 0; j < 8; ++j) { r[j] = lo[j]; r[8 + j] = hi[j]; }
  return r;
}

// block-wide reduction of two running sums (sum, sumsq). nthreads multiple of 32.
__device__ __forceinline__ void block_reduce2(float& a, float& b, float* sh, int tid, int nthreads) {
  #pragma unroll
  for (int off = 16; off > 0; off >>= 1) {
    a += __shfl_down(a, off, 32);
    b += __shfl_down(b, off, 32);
  }
  int wid = tid >> 5, lane = tid & 31, nw = nthreads >> 5;
  if (lane == 0) { sh[wid * 2] = a; sh[wid * 2 + 1] = b; }
  __syncthreads();
  float sa = 0.f, sb = 0.f;
  for (int i = 0; i < nw; ++i) { sa += sh[i * 2]; sb += sh[i * 2 + 1]; }
  __syncthreads();
  a = sa; b = sb;
}

// ---------------- weight f32 -> bf16 conversion ----------------
__global__ __launch_bounds__(256)
void cvt_bf16_kernel(const float* __restrict__ in, bf16_t* __restrict__ out, int n) {
  int i = blockIdx.x * 256 + threadIdx.x;
  if (i < n) out[i] = (bf16_t)in[i];
}

// ---------------- K1: LN1 then blk_ln (fused); h emitted as bf16 ----------------
__global__ __launch_bounds__(512)
void ln1_kernel(const float* __restrict__ x, const float* __restrict__ w1,
                const float* __restrict__ b1, const float* __restrict__ blkw,
                float* __restrict__ xn, bf16_t* __restrict__ hbf) {
  int row = blockIdx.x, i = threadIdx.x;
  __shared__ float sh[32];
  float v = x[(size_t)row * D_ + i];
  float a = v, b = v * v;
  block_reduce2(a, b, sh, i, D_);
  float m = a / D_, var = b / D_ - m * m;
  float xv = (v - m) * rsqrtf(var + 1e-5f) * w1[i] + b1[i];
  xn[(size_t)row * D_ + i] = xv;
  float a2 = xv, c2 = xv * xv;
  block_reduce2(a2, c2, sh, i, D_);
  float m2 = a2 / D_, v2 = c2 / D_ - m2 * m2;
  hbf[(size_t)row * D_ + i] = (bf16_t)((xv - m2) * rsqrtf(v2 + 1e-5f) * blkw[i]);
}

// ---------------- WMMA bf16 GEMM, async-LDS double-buffered B, branch-free main loop ----------------
// C = act(A[M,K]bf16 @ W[N,K]bf16^T + bias) + resid ; outputs optional f32 and/or bf16
// grid: (N/64, M/128), block 128 (4 waves). Wave computes 32(M) x 64(N); B tile shared via LDS.
__global__ __launch_bounds__(128)
void wmma_gemm_kernel(const bf16_t* __restrict__ A, const bf16_t* __restrict__ W,
                      const float* __restrict__ bias, const float* __restrict__ resid,
                      float* __restrict__ Cf, bf16_t* __restrict__ Cbf,
                      int M, int N, int K, int act) {
  __shared__ __align__(64) bf16_t lw[2][64 * 32];   // two 4KB B-tile buffers
  int tid = threadIdx.x;
  int wave = tid >> 5, lane = tid & 31;
  int half = lane >> 4, lm = lane & 15;
  int m0 = blockIdx.y * 128 + wave * 32;
  int n0 = blockIdx.x * 64;

  // staging: thread covers 16 bf16 (32B): row nl = tid/2, half-row hh = tid&1
  int nl = tid >> 1, hh = tid & 1;
  const bf16_t* wsrc = W + (size_t)(n0 + nl) * K + hh * 16;
  uint32_t lds0 = (uint32_t)(uintptr_t)&lw[0][nl * 32 + hh * 16];
  uint32_t lds1 = (uint32_t)(uintptr_t)&lw[1][nl * 32 + hh * 16];

  v8f zero8 = {0.f, 0.f, 0.f, 0.f, 0.f, 0.f, 0.f, 0.f};
  v8f acc[2][4];
  #pragma unroll
  for (int mt = 0; mt < 2; ++mt)
    #pragma unroll
    for (int t = 0; t < 4; ++t) acc[mt][t] = zero8;

  const bf16_t* a0p = A + (size_t)(m0 + lm) * K;
  const bf16_t* a1p = A + (size_t)(m0 + 16 + lm) * K;

  auto stage = [&](int kb, uint32_t ld) {
    uint64_t g = (uint64_t)(uintptr_t)(wsrc + kb);
    asm volatile("global_load_async_to_lds_b128 %0, %1, off"
                 :: "v"(ld), "v"(g) : "memory");
    asm volatile("global_load_async_to_lds_b128 %0, %1, off offset:16"
                 :: "v"(ld), "v"(g) : "memory");
  };
  auto compute = [&](int kb, int buf) {
    v8bf lo0 = *(const v8bf*)(a0p + kb + 16 * half);
    v8bf hi0 = *(const v8bf*)(a0p + kb + 16 * half + 8);
    v8bf lo1 = *(const v8bf*)(a1p + kb + 16 * half);
    v8bf hi1 = *(const v8bf*)(a1p + kb + 16 * half + 8);
    v16bf am0 = join16(lo0, hi0);
    v16bf am1 = join16(lo1, hi1);
    const bf16_t* lwb = &lw[buf][0];
    #pragma unroll
    for (int t = 0; t < 4; ++t) {
      v16bf b = *(const v16bf*)&lwb[(t * 16 + lm) * 32 + 16 * half];
      acc[0][t] = __builtin_amdgcn_wmma_f32_16x16x32_bf16(false, am0, false, b,
                                                          (short)0, acc[0][t], false, false);
      acc[1][t] = __builtin_amdgcn_wmma_f32_16x16x32_bf16(false, am1, false, b,
                                                          (short)0, acc[1][t], false, false);
    }
  };

  // prologue: stage kb=0 into buf0
  stage(0, lds0);
  int buf = 0;
  int kb = 0;
  // branch-free steady state: always stage next + wait for current
  for (; kb + 32 < K; kb += 32) {
    stage(kb + 32, buf ? lds0 : lds1);
    asm volatile("s_wait_asynccnt 0x2" ::: "memory");   // current tile landed
    __syncthreads();
    compute(kb, buf);
    __syncthreads();   // protect buffer before restaging
    buf ^= 1;
  }
  // peeled last iteration
  asm volatile("s_wait_asynccnt 0x0" ::: "memory");
  __syncthreads();
  compute(kb, buf);

  #pragma unroll
  for (int mt = 0; mt < 2; ++mt) {
    #pragma unroll
    for (int t = 0; t < 4; ++t) {
      #pragma unroll
      for (int r = 0; r < 8; ++r) {
        int m = m0 + mt * 16 + r + 8 * half;
        int n = n0 + t * 16 + lm;
        float v = acc[mt][t][r];
        if (bias)  v += bias[n];
        if (act == 1) v = silu_f(v);
        if (resid) v += resid[(size_t)m * N + n];
        if (Cf)  Cf[(size_t)m * N + n] = v;
        if (Cbf) Cbf[(size_t)m * N + n] = (bf16_t)v;
      }
    }
  }
}

// ---------------- K3: depthwise causal conv (KS=4) + SiLU ----------------
__global__ __launch_bounds__(256)
void conv_silu_kernel(const float* __restrict__ up, const float* __restrict__ cw,
                      const float* __restrict__ cb, float* __restrict__ xc) {
  size_t idx = (size_t)blockIdx.x * blockDim.x + threadIdx.x;
  if (idx >= (size_t)ROWS_ * INNER_) return;
  int c = (int)(idx & (INNER_ - 1));
  int row = (int)(idx >> 10);
  int s = row & (S_ - 1);
  int b = row >> 10;
  float acc = cb[c];
  #pragma unroll
  for (int t = 0; t < 4; ++t) {
    int ss = s - 3 + t;
    if (ss >= 0) acc += cw[c * 4 + t] * up[((size_t)(b * S_ + ss)) * (2 * INNER_) + c];
  }
  xc[idx] = silu_f(acc);
}

// ---------------- K4: head-wise 4x4 q/k/v -> bf16 (+ v transposed for attention) ----------------
__global__ __launch_bounds__(256)
void headwise_qkv_kernel(const float* __restrict__ xc, const float* __restrict__ up,
                         const float* __restrict__ qw, const float* __restrict__ kw,
                         const float* __restrict__ vw,
                         bf16_t* __restrict__ q, bf16_t* __restrict__ k,
                         bf16_t* __restrict__ v, bf16_t* __restrict__ vT) {
  size_t idx = (size_t)blockIdx.x * blockDim.x + threadIdx.x;
  if (idx >= (size_t)ROWS_ * INNER_) return;
  int i = (int)(idx & (INNER_ - 1));
  int row = (int)(idx >> 10);
  int n = i >> 2, o = i & 3;
  const float* xcp = xc + (size_t)row * INNER_ + n * 4;
  const float* xmp = up + (size_t)row * (2 * INNER_) + n * 4;  // x_m = up[..., :INNER]
  float aq = 0.f, ak = 0.f, av = 0.f;
  #pragma unroll
  for (int d = 0; d < 4; ++d) {
    aq += qw[(n * 4 + o) * 4 + d] * xcp[d];
    ak += kw[(n * 4 + o) * 4 + d] * xcp[d];
    av += vw[(n * 4 + o) * 4 + d] * xmp[d];
  }
  q[idx] = (bf16_t)aq;
  k[idx] = (bf16_t)ak;
  v[idx] = (bf16_t)av;
  int b = row >> 10, s = row & (S_ - 1);
  int h = i >> 8, dl = i & (DH_ - 1);
  vT[((size_t)(b * NH_ + h) * DH_ + dl) * S_ + s] = (bf16_t)av;
}

// ---------------- K5: gate pre-activations (GEMV, wave per head) ----------------
__global__ __launch_bounds__(128)
void gates_kernel(const bf16_t* __restrict__ q, const bf16_t* __restrict__ k,
                  const bf16_t* __restrict__ v, const float* __restrict__ igw,
                  const float* __restrict__ igb, const float* __restrict__ fgw,
                  const float* __restrict__ fgb,
                  float* __restrict__ ipre, float* __restrict__ fpre) {
  int row = blockIdx.x;
  int h = threadIdx.x >> 5, lane = threadIdx.x & 31;
  const bf16_t* qb = q + (size_t)row * INNER_;
  const bf16_t* kb = k + (size_t)row * INNER_;
  const bf16_t* vb = v + (size_t)row * INNER_;
  const float* iw = igw + (size_t)h * (3 * INNER_);
  const float* fw = fgw + (size_t)h * (3 * INNER_);
  float ai = 0.f, af = 0.f;
  for (int j = lane; j < INNER_; j += 32) {
    float a0 = (float)qb[j], a1 = (float)kb[j], a2 = (float)vb[j];
    ai += iw[j] * a0 + iw[INNER_ + j] * a1 + iw[2 * INNER_ + j] * a2;
    af += fw[j] * a0 + fw[INNER_ + j] * a1 + fw[2 * INNER_ + j] * a2;
  }
  #pragma unroll
  for (int off = 16; off > 0; off >>= 1) {
    ai += __shfl_down(ai, off, 32);
    af += __shfl_down(af, off, 32);
  }
  if (lane == 0) {
    int b = row >> 10, s = row & (S_ - 1);
    size_t o = ((size_t)(b * NH_ + h)) * S_ + s;
    ipre[o] = ai + igb[h];
    fpre[o] = af + fgb[h];
  }
}

// ---------------- K6a: gate scan: C = cumsum(log_sigmoid(f)); g = i - C; mg = prefixmax(g) ----------------
__global__ __launch_bounds__(1024)
void gate_scan_kernel(const float* __restrict__ ipre, const float* __restrict__ fpre,
                      float* __restrict__ g, float* __restrict__ mg, float* __restrict__ fl) {
  int bh = blockIdx.x;
  int s = threadIdx.x;
  __shared__ float sh[S_];
  float f = fpre[(size_t)bh * S_ + s];
  float lf = fminf(f, 0.f) - log1pf(__expf(-fabsf(f)));   // stable log_sigmoid
  sh[s] = lf;
  __syncthreads();
  for (int off = 1; off < S_; off <<= 1) {
    float v = (s >= off) ? sh[s - off] : 0.f;
    __syncthreads();
    sh[s] += v;
    __syncthreads();
  }
  float Cc = sh[s];                       // inclusive cumsum == cs[s+1]
  float gv = ipre[(size_t)bh * S_ + s] - Cc;
  g[(size_t)bh * S_ + s] = gv;
  __syncthreads();
  sh[s] = gv;
  __syncthreads();
  for (int off = 1; off < S_; off <<= 1) {
    float v = (s >= off) ? sh[s - off] : -INFINITY;
    __syncthreads();
    sh[s] = fmaxf(sh[s], v);
    __syncthreads();
  }
  float mgv = sh[s];                      // max_log_d = Cc + mgv
  mg[(size_t)bh * S_ + s] = mgv;
  fl[(size_t)bh * S_ + s] = __expf(-Cc - mgv);
}

// ---------------- K6b: single-pass flash mLSTM (wave per 16-query block) ----------------
__global__ __launch_bounds__(32)
void mlstm_attn_kernel(const bf16_t* __restrict__ q, const bf16_t* __restrict__ kk,
                       const bf16_t* __restrict__ vT, const float* __restrict__ g,
                       const float* __restrict__ mg, const float* __restrict__ fl,
                       float* __restrict__ hcell) {
  int rb = blockIdx.x;        // 0..63
  int bh = blockIdx.y;        // 0..15
  int b = bh >> 2, h = bh & 3;
  int lane = threadIdx.x;
  int half = lane >> 4, lm = lane & 15;

  __shared__ __align__(64) bf16_t cl[16][32];

  const bf16_t* qbase = q  + ((size_t)(b * S_)) * INNER_ + h * DH_;
  const bf16_t* kbase = kk + ((size_t)(b * S_)) * INNER_ + h * DH_;
  const bf16_t* vbase = vT + (size_t)bh * DH_ * S_;
  const float* gbh  = g  + (size_t)bh * S_;
  const float* mgbh = mg + (size_t)bh * S_;
  const float* flbh = fl + (size_t)bh * S_;

  int s0 = rb * 16;

  // q fragments: 8 A-frags over K = DH = 256 (contiguous 16B bf16 loads)
  v16bf qf[8];
  {
    const bf16_t* qp = qbase + (size_t)(s0 + lm) * INNER_;
    #pragma unroll
    for (int kt = 0; kt < 8; ++kt) {
      v8bf lo = *(const v8bf*)(qp + kt * 32 + 16 * half);
      v8bf hi = *(const v8bf*)(qp + kt * 32 + 16 * half + 8);
      qf[kt] = join16(lo, hi);
    }
  }
  float mgv[8];
  #pragma unroll
  for (int r = 0; r < 8; ++r) mgv[r] = mgbh[s0 + r + 8 * half];

  v8f zero8 = {0.f, 0.f, 0.f, 0.f, 0.f, 0.f, 0.f, 0.f};
  v8f out[16];
  #pragma unroll
  for (int t = 0; t < 16; ++t) out[t] = zero8;
  float rs[8] = {};

  const float scale = 0.0625f;  // DH^-0.5

  for (int cp = 0; cp <= (rb >> 1); ++cp) {
    #pragma unroll
    for (int sub = 0; sub < 2; ++sub) {
      int t0 = cp * 32 + sub * 16;
      if (t0 <= s0 + 15) {
        v8f sacc = zero8;
        const bf16_t* kp = kbase + (size_t)(t0 + lm) * INNER_;
        #pragma unroll
        for (int grp = 0; grp < 2; ++grp) {          // batch 4 loads per wait
          v16bf kf[4];
          #pragma unroll
          for (int u = 0; u < 4; ++u)
            kf[u] = *(const v16bf*)(kp + (grp * 4 + u) * 32 + 16 * half);
          #pragma unroll
          for (int u = 0; u < 4; ++u)
            sacc = __builtin_amdgcn_wmma_f32_16x16x32_bf16(false, qf[grp * 4 + u], false,
                                                           kf[u], (short)0, sacc, false, false);
        }
        int t = t0 + lm;
        float gt = gbh[t];
        #pragma unroll
        for (int r = 0; r < 8; ++r) {
          int s = s0 + r + 8 * half;
          float val = 0.f;
          if (t <= s) val = sacc[r] * scale * __expf(gt - mgv[r]);
          rs[r] += val;
          cl[r + 8 * half][sub * 16 + lm] = (bf16_t)val;
        }
      } else {
        #pragma unroll
        for (int r = 0; r < 8; ++r) cl[r + 8 * half][sub * 16 + lm] = (bf16_t)0.f;
      }
    }
    __syncthreads();
    // re-fragment c tile (C layout -> A layout) via LDS
    v8bf clo = *(const v8bf*)&cl[lm][16 * half];
    v8bf chi = *(const v8bf*)&cl[lm][16 * half + 8];
    v16bf af = join16(clo, chi);
    __syncthreads();
    // out += c_tile @ v_slab ; vT rows are contiguous in t
    #pragma unroll
    for (int ntg = 0; ntg < 4; ++ntg) {
      v16bf vf[4];
      #pragma unroll
      for (int u = 0; u < 4; ++u) {
        int d = (ntg * 4 + u) * 16 + lm;
        vf[u] = *(const v16bf*)(vbase + (size_t)d * S_ + cp * 32 + 16 * half);
      }
      #pragma unroll
      for (int u = 0; u < 4; ++u)
        out[ntg * 4 + u] = __builtin_amdgcn_wmma_f32_16x16x32_bf16(false, af, false, vf[u],
                                                                   (short)0, out[ntg * 4 + u],
                                                                   false, false);
    }
  }

  #pragma unroll
  for (int r = 0; r < 8; ++r) {
    float v_ = rs[r];
    v_ += __shfl_xor(v_, 1, 32);
    v_ += __shfl_xor(v_, 2, 32);
    v_ += __shfl_xor(v_, 4, 32);
    v_ += __shfl_xor(v_, 8, 32);
    rs[r] = v_;
  }
  float inv[8];
  #pragma unroll
  for (int r = 0; r < 8; ++r) {
    int s = s0 + r + 8 * half;
    float norm = fmaxf(fabsf(rs[r]), flbh[s]);
    inv[r] = 1.0f / (norm + 1e-6f);
  }
  #pragma unroll
  for (int nt = 0; nt < 16; ++nt) {
    #pragma unroll
    for (int r = 0; r < 8; ++r) {
      int s = s0 + r + 8 * half;
      int d = nt * 16 + lm;
      hcell[((size_t)(b * S_ + s)) * INNER_ + h * DH_ + d] = out[nt][r] * inv[r];
    }
  }
}

// ---------------- K7: per-head norm + skip + SiLU(z) gate -> hs (bf16) ----------------
__global__ __launch_bounds__(1024)
void mhnorm_gate_kernel(const float* __restrict__ up, const float* __restrict__ xc,
                        const float* __restrict__ hc, const float* __restrict__ onw,
                        const float* __restrict__ skw, bf16_t* __restrict__ hsbf) {
  int row = blockIdx.x;
  int i = threadIdx.x;               // 0..1023
  int h = i >> 8;
  __shared__ float sh[64];
  float v = hc[(size_t)row * INNER_ + i];
  float a = v, b = v * v;
  #pragma unroll
  for (int off = 16; off > 0; off >>= 1) {
    a += __shfl_down(a, off, 32);
    b += __shfl_down(b, off, 32);
  }
  int wid = i >> 5, lane = i & 31;
  if (lane == 0) { sh[wid * 2] = a; sh[wid * 2 + 1] = b; }
  __syncthreads();
  float sa = 0.f, sb = 0.f;
  for (int w = h * 8; w < h * 8 + 8; ++w) { sa += sh[w * 2]; sb += sh[w * 2 + 1]; }
  __syncthreads();
  float m = sa / (float)DH_, var = sb / (float)DH_ - m * m;
  float hn = (v - m) * rsqrtf(var + 1e-5f) * onw[i];
  float z = up[(size_t)row * (2 * INNER_) + INNER_ + i];
  float hs = (hn + skw[i] * xc[(size_t)row * INNER_ + i]) * silu_f(z);
  hsbf[(size_t)row * INNER_ + i] = (bf16_t)hs;
}

// ---------------- K8: post_norm residual + LN2 (fused); mv emitted bf16 ----------------
__global__ __launch_bounds__(512)
void post_ln_kernel(const float* __restrict__ y, const float* __restrict__ x,
                    const float* __restrict__ pw, const float* __restrict__ w2,
                    const float* __restrict__ b2, float* __restrict__ x2,
                    bf16_t* __restrict__ mvbf) {
  int row = blockIdx.x, i = threadIdx.x;
  __shared__ float sh[32];
  float yv = y[(size_t)row * D_ + i];
  float a = yv, b = yv * yv;
  block_reduce2(a, b, sh, i, D_);
  float m = a / D_, var = b / D_ - m * m;
  float xl = (yv - m) * rsqrtf(var + 1e-5f) * pw[i];
  float xv2 = x[(size_t)row * D_ + i] + xl;
  x2[(size_t)row * D_ + i] = xv2;
  float a2 = xv2, c2 = xv2 * xv2;
  block_reduce2(a2, c2, sh, i, D_);
  float m2 = a2 / D_, v2 = c2 / D_ - m2 * m2;
  mvbf[(size_t)row * D_ + i] = (bf16_t)((xv2 - m2) * rsqrtf(v2 + 1e-5f) * w2[i] + b2[i]);
}

// ---------------- launch ----------------
extern "C" void kernel_launch(void* const* d_in, const int* in_sizes, int n_in,
                              void* d_out, int out_size, void* d_ws, size_t ws_size,
                              hipStream_t stream) {
  const float* x      = (const float*)d_in[0];
  const float* ln1_w  = (const float*)d_in[1];
  const float* ln1_b  = (const float*)d_in[2];
  const float* ln2_w  = (const float*)d_in[3];
  const float* ln2_b  = (const float*)d_in[4];
  const float* blk_ln = (const float*)d_in[5];
  const float* pu_w   = (const float*)d_in[6];
  const float* conv_w = (const float*)d_in[7];
  const float* conv_b = (const float*)d_in[8];
  const float* q_w    = (const float*)d_in[9];
  const float* k_w    = (const float*)d_in[10];
  const float* v_w    = (const float*)d_in[11];
  const float* ig_w   = (const float*)d_in[12];
  const float* ig_b   = (const float*)d_in[13];
  const float* fg_w   = (const float*)d_in[14];
  const float* fg_b   = (const float*)d_in[15];
  const float* on_w   = (const float*)d_in[16];
  const float* sk_w   = (const float*)d_in[17];
  const float* pd_w   = (const float*)d_in[18];
  const float* pn_w   = (const float*)d_in[19];
  const float* m_w1   = (const float*)d_in[20];
  const float* m_b1   = (const float*)d_in[21];
  const float* m_w2   = (const float*)d_in[22];
  const float* m_b2   = (const float*)d_in[23];
  const float* m_w3   = (const float*)d_in[24];
  const float* m_b3   = (const float*)d_in[25];
  float* out = (float*)d_out;

  // ---- workspace layout (byte bump allocator) ----
  char* p = (char*)d_ws;
  auto alloc = [&](size_t bytes) -> char* {
    char* r = p;
    p += (bytes + 255) & ~(size_t)255;
    return r;
  };
  const size_t RD = (size_t)ROWS_ * D_;        // 2,097,152 elems
  const size_t RI = (size_t)ROWS_ * INNER_;    // 4,194,304 elems

  float*  xn   = (float*)alloc(RD * 4);
  float*  y    = (float*)alloc(RD * 4);
  float*  up   = (float*)alloc(RI * 2 * 4);
  float*  xc   = (float*)alloc(RI * 4);
  bf16_t* qbf  = (bf16_t*)alloc(RI * 2);       // |-- region reused by t1bf/t2bf
  bf16_t* kbf  = (bf16_t*)alloc(RI * 2);
  bf16_t* vbf  = (bf16_t*)alloc(RI * 2);
  bf16_t* vT   = (bf16_t*)alloc(RI * 2);       // --|
  float*  hc   = (float*)alloc(RI * 4);
  bf16_t* hsbf = (bf16_t*)alloc(RI * 2);
  float*  x2   = (float*)alloc(RD * 4);
  bf16_t* mvbf = (bf16_t*)alloc(RD * 2);
  bf16_t* hbf  = (bf16_t*)alloc(RD * 2);
  bf16_t* wpu  = (bf16_t*)alloc((size_t)(2 * INNER_) * D_ * 2);
  bf16_t* wpd  = (bf16_t*)alloc((size_t)D_ * INNER_ * 2);
  bf16_t* w1b  = (bf16_t*)alloc((size_t)MLP_H_ * D_ * 2);
  bf16_t* w2b  = (bf16_t*)alloc((size_t)MLP_H_ * MLP_H_ * 2);
  bf16_t* w3b  = (bf16_t*)alloc((size_t)D_ * MLP_H_ * 2);
  const size_t GS = (size_t)B_ * NH_ * S_;
  float* ip  = (float*)alloc(GS * 4);
  float* fp  = (float*)alloc(GS * 4);
  float* gg  = (float*)alloc(GS * 4);
  float* mgp = (float*)alloc(GS * 4);
  float* flp = (float*)alloc(GS * 4);
  bf16_t* t1bf = qbf;   // 8.4M bf16 spans qbf+kbf (free after attention/gates)
  bf16_t* t2bf = vbf;   // spans vbf+vT

  // ---- weight conversions (bf16) ----
  auto cvt = [&](const float* src, bf16_t* dst, int n) {
    cvt_bf16_kernel<<<(n + 255) / 256, 256, 0, stream>>>(src, dst, n);
  };
  cvt(pu_w, wpu, 2 * INNER_ * D_);
  cvt(pd_w, wpd, D_ * INNER_);
  cvt(m_w1, w1b, MLP_H_ * D_);
  cvt(m_w2, w2b, MLP_H_ * MLP_H_);
  cvt(m_w3, w3b, D_ * MLP_H_);

  // 1. LN1 + blk_ln
  ln1_kernel<<<ROWS_, D_, 0, stream>>>(x, ln1_w, ln1_b, blk_ln, xn, hbf);

  // 2. up = h @ proj_up_w^T   (M=4096, N=2048, K=512)
  wmma_gemm_kernel<<<dim3(2 * INNER_ / 64, ROWS_ / 128), 128, 0, stream>>>(
      hbf, wpu, nullptr, nullptr, up, nullptr, ROWS_, 2 * INNER_, D_, 0);

  // 3. causal depthwise conv + SiLU
  conv_silu_kernel<<<(ROWS_ * INNER_ + 255) / 256, 256, 0, stream>>>(up, conv_w, conv_b, xc);

  // 4. head-wise q/k/v (bf16 + v transposed)
  headwise_qkv_kernel<<<(ROWS_ * INNER_ + 255) / 256, 256, 0, stream>>>(
      xc, up, q_w, k_w, v_w, qbf, kbf, vbf, vT);

  // 5. gate pre-activations
  gates_kernel<<<ROWS_, 128, 0, stream>>>(qbf, kbf, vbf, ig_w, ig_b, fg_w, fg_b, ip, fp);

  // 6. per-(b,h) gate scan
  gate_scan_kernel<<<B_ * NH_, S_, 0, stream>>>(ip, fp, gg, mgp, flp);

  // 7. single-pass flash mLSTM
  mlstm_attn_kernel<<<dim3(S_ / 16, B_ * NH_), 32, 0, stream>>>(qbf, kbf, vT, gg, mgp, flp, hc);

  // 8. per-head norm + skip + SiLU(z) gate -> hs (bf16)
  mhnorm_gate_kernel<<<ROWS_, INNER_, 0, stream>>>(up, xc, hc, on_w, sk_w, hsbf);

  // 9. y = xn + hs @ proj_down_w^T   (M=4096, N=512, K=1024)
  wmma_gemm_kernel<<<dim3(D_ / 64, ROWS_ / 128), 128, 0, stream>>>(
      hsbf, wpd, nullptr, xn, y, nullptr, ROWS_, D_, INNER_, 0);

  // 10. x2 = x + LN(y)*post_w ; mv = LN2(x2) (bf16)
  post_ln_kernel<<<ROWS_, D_, 0, stream>>>(y, x, pn_w, ln2_w, ln2_b, x2, mvbf);

  // 11. MLP
  wmma_gemm_kernel<<<dim3(MLP_H_ / 64, ROWS_ / 128), 128, 0, stream>>>(
      mvbf, w1b, m_b1, nullptr, nullptr, t1bf, ROWS_, MLP_H_, D_, 1);
  wmma_gemm_kernel<<<dim3(MLP_H_ / 64, ROWS_ / 128), 128, 0, stream>>>(
      t1bf, w2b, m_b2, nullptr, nullptr, t2bf, ROWS_, MLP_H_, MLP_H_, 1);
  wmma_gemm_kernel<<<dim3(D_ / 64, ROWS_ / 128), 128, 0, stream>>>(
      t2bf, w3b, m_b3, x2, out, nullptr, ROWS_, D_, MLP_H_, 0);
}
